// CrossAttention_70222715290160
// MI455X (gfx1250) — matrix operand order repaired and checked by reading, not compiled, not run on hardware
//
#include <hip/hip_runtime.h>

// ---------------------------------------------------------------------------
// Shapes (fixed by the reference)
// ---------------------------------------------------------------------------
#define BB   16
#define NN   256
#define TT   4096
#define DD   1024
#define HH   8
#define DH   128
#define SCALE 0.08838834764831845f   // 128^-0.5

typedef unsigned short u16;
typedef __attribute__((ext_vector_type(16))) __bf16 bf16x16;
typedef __attribute__((ext_vector_type(8)))  float  f32x8;

union ABFrag { bf16x16 v; unsigned u[8]; };

// round-to-nearest-even f32 -> bf16
static __device__ inline u16 f2bf(float f) {
    union { float f; unsigned u; } a; a.f = f;
    unsigned u = a.u;
    unsigned r = u + 0x7FFFu + ((u >> 16) & 1u);
    if ((u & 0x7F800000u) == 0x7F800000u) r = u;   // inf/nan passthrough
    return (u16)(r >> 16);
}

static __device__ inline f32x8 wmma_bf16(const ABFrag& a, const ABFrag& b, f32x8 c) {
    return __builtin_amdgcn_wmma_f32_16x16x32_bf16(
        false, a.v, false, b.v, (short)0, c, false, false);
}

#define ZERO8 f32x8{0.f,0.f,0.f,0.f,0.f,0.f,0.f,0.f}

// ---------------------------------------------------------------------------
// Kernel 1: row LayerNorm (row length 1024) -> bf16, optional scale folded in
// ---------------------------------------------------------------------------
__global__ __launch_bounds__(256)
void ln_bf16_kernel(const float* __restrict__ in, const float* __restrict__ gamma,
                    u16* __restrict__ out, float scale)
{
    const int row = blockIdx.x;
    const int tid = threadIdx.x;
    const float* p = in + (size_t)row * DD;

    float v[4], s = 0.f, s2 = 0.f;
#pragma unroll
    for (int j = 0; j < 4; ++j) {
        v[j] = p[tid + j * 256];
        s  += v[j];
        s2 += v[j] * v[j];
    }
#pragma unroll
    for (int m = 1; m < 32; m <<= 1) {
        s  += __shfl_xor(s,  m, 32);
        s2 += __shfl_xor(s2, m, 32);
    }
    __shared__ float rs[8], rq[8];
    if ((tid & 31) == 0) { rs[tid >> 5] = s; rq[tid >> 5] = s2; }
    __syncthreads();
    float ts = 0.f, t2 = 0.f;
#pragma unroll
    for (int w = 0; w < 8; ++w) { ts += rs[w]; t2 += rq[w]; }

    const float mean = ts * (1.f / DD);
    const float var  = t2 * (1.f / DD) - mean * mean;
    const float rstd = rsqrtf(var + 1e-5f);

    u16* o = out + (size_t)row * DD;
#pragma unroll
    for (int j = 0; j < 4; ++j) {
        const int c = tid + j * 256;
        o[c] = f2bf((v[j] - mean) * rstd * gamma[c] * scale);
    }
}

// ---------------------------------------------------------------------------
// Kernel 2: W_kv[:, :128] (f32, 1024x256) -> Wk_t bf16 [n=128][d=1024]
// ---------------------------------------------------------------------------
__global__ __launch_bounds__(256)
void wkt_kernel(const float* __restrict__ wkv, u16* __restrict__ wkt)
{
    const int idx = blockIdx.x * 256 + threadIdx.x;   // 0 .. 131071
    const int n = idx >> 10;
    const int d = idx & 1023;
    wkt[idx] = f2bf(wkv[d * 256 + n]);                // wkt[n*1024 + d]
}

// ---------------------------------------------------------------------------
// Kernel 3: tiled transpose  xt[b][n][t] = xnb[b][t][n]   (bf16)
// grid = B*(T/64)*(D/64) = 16384; block 256. LDS-tiled, vector loads/stores.
// ---------------------------------------------------------------------------
__global__ __launch_bounds__(256)
void xt_kernel(const u16* __restrict__ xnb, u16* __restrict__ xt)
{
    const int bid = blockIdx.x;
    const int nb  = bid & 15;            // D/64
    const int tmp = bid >> 4;
    const int tb  = tmp & 63;            // T/64
    const int b   = tmp >> 6;
    const int t0  = tb * 64, n0 = nb * 64;

    __shared__ u16 Tl[64][72];           // +8 pad vs bank conflicts
    const int tid = threadIdx.x;

#pragma unroll
    for (int it = 0; it < 2; ++it) {
        const int v  = tid + it * 256;   // 0..511 vectors of 8 bf16
        const int t  = v >> 3;
        const int nv = (v & 7) * 8;
        const uint4 d = *(const uint4*)(xnb + (size_t)(b * TT + t0 + t) * DD + n0 + nv);
        Tl[nv + 0][t] = (u16)(d.x);  Tl[nv + 1][t] = (u16)(d.x >> 16);
        Tl[nv + 2][t] = (u16)(d.y);  Tl[nv + 3][t] = (u16)(d.y >> 16);
        Tl[nv + 4][t] = (u16)(d.z);  Tl[nv + 5][t] = (u16)(d.z >> 16);
        Tl[nv + 6][t] = (u16)(d.w);  Tl[nv + 7][t] = (u16)(d.w >> 16);
    }
    __syncthreads();
#pragma unroll
    for (int it = 0; it < 2; ++it) {
        const int v  = tid + it * 256;
        const int n  = v >> 3;
        const int tv = (v & 7) * 8;
        uint4 o;
        o.x = (unsigned)Tl[n][tv + 0] | ((unsigned)Tl[n][tv + 1] << 16);
        o.y = (unsigned)Tl[n][tv + 2] | ((unsigned)Tl[n][tv + 3] << 16);
        o.z = (unsigned)Tl[n][tv + 4] | ((unsigned)Tl[n][tv + 5] << 16);
        o.w = (unsigned)Tl[n][tv + 6] | ((unsigned)Tl[n][tv + 7] << 16);
        *(uint4*)(xt + (size_t)(b * DD + n0 + n) * TT + t0 + tv) = o;
    }
}

// ---------------------------------------------------------------------------
// Kernel 4: k[b,t,0:128] = x_nb[b,t,:] @ Wk_t  (bf16 WMMA, f32 accum -> bf16)
// grid = B * (T/128); block = 256 (8 waves). Each wave: 16 rows x 128 cols.
// ---------------------------------------------------------------------------
__global__ __launch_bounds__(256)
void kproj_kernel(const u16* __restrict__ xnb, const u16* __restrict__ wkt,
                  u16* __restrict__ kbuf)
{
    const int bid = blockIdx.x;
    const int b  = bid >> 5;
    const int mt = bid & 31;

    const int tid  = threadIdx.x;
    const int w    = tid >> 5;
    const int l    = tid & 31;
    const int ln   = l & 15;
    const int half = l >> 4;

    const int m0 = mt * 128 + w * 16;
    const u16* arow = xnb + ((size_t)(b * TT + m0 + ln)) * DD;

    f32x8 acc[8];
#pragma unroll
    for (int t = 0; t < 8; ++t) acc[t] = ZERO8;

    for (int ks = 0; ks < DD; ks += 32) {
        ABFrag a;
#pragma unroll
        for (int i = 0; i < 8; ++i) {
            const int k0 = ks + half * 16 + i * 2;
            a.u[i] = *(const unsigned*)(arow + k0);
        }
#pragma unroll
        for (int g = 0; g < 2; ++g) {
            ABFrag bf[4];
#pragma unroll
            for (int j = 0; j < 4; ++j) {
                const u16* wp = wkt + (size_t)((g * 4 + j) * 16 + ln) * DD + ks + half * 16;
#pragma unroll
                for (int i = 0; i < 8; ++i) bf[j].u[i] = *(const unsigned*)(wp + 2 * i);
            }
#pragma unroll
            for (int j = 0; j < 4; ++j)
                acc[g * 4 + j] = wmma_bf16(a, bf[j], acc[g * 4 + j]);
        }
    }
#pragma unroll
    for (int nt = 0; nt < 8; ++nt)
#pragma unroll
        for (int r = 0; r < 8; ++r)
            kbuf[((size_t)(b * TT + m0 + half * 8 + r)) * DH + nt * 16 + ln] =
                f2bf(acc[nt][r]);
}

// ---------------------------------------------------------------------------
// Kernel 5: sim = (q*scale) @ k^T  + full softmax, attn (f32) -> d_out
// grid = B*H*(N/16) = 2048; block = 1024 (32 waves, each owns 128 cols of T).
// Entire 16x4096 score row-block stays in registers.
// ---------------------------------------------------------------------------
__global__ __launch_bounds__(1024)
void sim_softmax_kernel(const u16* __restrict__ qnb, const u16* __restrict__ kbuf,
                        float* __restrict__ attn)
{
    const int bid  = blockIdx.x;
    const int b    = bid >> 7;
    const int r128 = bid & 127;
    const int h    = r128 >> 4;
    const int mb   = r128 & 15;
    const int n0   = mb * 16;

    const int tid  = threadIdx.x;
    const int w    = tid >> 5;      // 0..31
    const int l    = tid & 31;
    const int ln   = l & 15;
    const int half = l >> 4;

    const u16* qbase = qnb + ((size_t)(b * NN + n0 + ln)) * DD + h * DH;
    const u16* kb    = kbuf + (size_t)b * TT * DH;
    const int colbase = w * 128;

    f32x8 acc[8];
#pragma unroll
    for (int t = 0; t < 8; ++t) acc[t] = ZERO8;

    for (int ks = 0; ks < DH; ks += 32) {
        ABFrag a;
#pragma unroll
        for (int i = 0; i < 8; ++i) {
            const int k0 = ks + half * 16 + i * 2;
            a.u[i] = *(const unsigned*)(qbase + k0);
        }
#pragma unroll
        for (int g = 0; g < 2; ++g) {
            ABFrag bf[4];
#pragma unroll
            for (int j = 0; j < 4; ++j) {
                const int t = colbase + (g * 4 + j) * 16 + ln;
                const u16* kp = kb + (size_t)t * DH + ks + half * 16;
#pragma unroll
                for (int i = 0; i < 8; ++i) bf[j].u[i] = *(const unsigned*)(kp + 2 * i);
            }
#pragma unroll
            for (int j = 0; j < 4; ++j)
                acc[g * 4 + j] = wmma_bf16(a, bf[j], acc[g * 4 + j]);
        }
    }

    __shared__ float red[32][16];
    __shared__ float rowstat[16];

    // --- row max over all 4096 columns ---
    float vmax[8];
#pragma unroll
    for (int r = 0; r < 8; ++r) {
        vmax[r] = acc[0][r];
#pragma unroll
        for (int nt = 1; nt < 8; ++nt) vmax[r] = fmaxf(vmax[r], acc[nt][r]);
    }
#pragma unroll
    for (int m = 1; m < 16; m <<= 1)
#pragma unroll
        for (int r = 0; r < 8; ++r)
            vmax[r] = fmaxf(vmax[r], __shfl_xor(vmax[r], m, 32));
    if (ln == 0)
#pragma unroll
        for (int r = 0; r < 8; ++r) red[w][half * 8 + r] = vmax[r];
    __syncthreads();
    if (tid < 16) {
        float m = red[0][tid];
        for (int ww = 1; ww < 32; ++ww) m = fmaxf(m, red[ww][tid]);
        rowstat[tid] = m;
    }
    __syncthreads();
    float rmax[8];
#pragma unroll
    for (int r = 0; r < 8; ++r) rmax[r] = rowstat[half * 8 + r];

    // --- exp + row sum ---
    float vsum[8] = {0.f,0.f,0.f,0.f,0.f,0.f,0.f,0.f};
#pragma unroll
    for (int nt = 0; nt < 8; ++nt)
#pragma unroll
        for (int r = 0; r < 8; ++r) {
            const float e = __expf(acc[nt][r] - rmax[r]);
            acc[nt][r] = e;
            vsum[r] += e;
        }
#pragma unroll
    for (int m = 1; m < 16; m <<= 1)
#pragma unroll
        for (int r = 0; r < 8; ++r) vsum[r] += __shfl_xor(vsum[r], m, 32);
    __syncthreads();
    if (ln == 0)
#pragma unroll
        for (int r = 0; r < 8; ++r) red[w][half * 8 + r] = vsum[r];
    __syncthreads();
    if (tid < 16) {
        float s = 0.f;
        for (int ww = 0; ww < 32; ++ww) s += red[ww][tid];
        rowstat[tid] = 1.f / s;
    }
    __syncthreads();
    float rinv[8];
#pragma unroll
    for (int r = 0; r < 8; ++r) rinv[r] = rowstat[half * 8 + r];

    // --- write attn (f32) ---
    float* arow = attn + ((size_t)((b * HH + h) * NN + n0)) * TT;
#pragma unroll
    for (int nt = 0; nt < 8; ++nt) {
        const int t = colbase + nt * 16 + ln;
#pragma unroll
        for (int r = 0; r < 8; ++r)
            arow[(size_t)(half * 8 + r) * TT + t] = acc[nt][r] * rinv[r];
    }
}

// ---------------------------------------------------------------------------
// Kernel 6: out[b,n,h,:] = attn[b,h,n,:] @ x_nb[b,:,:]
// grid = B*H*(N/16) = 2048; block = 256 (8 waves, each owns 128 of 1024 cols).
// No LDS, no barriers: A built in-register from attn f32 (L0-shared across
// waves), B is straight b128 loads from pre-transposed xt (K-contiguous).
// ---------------------------------------------------------------------------
__global__ __launch_bounds__(256)
void out_gemm_kernel(const float* __restrict__ attn, const u16* __restrict__ xt,
                     float* __restrict__ out)
{
    const int bid = blockIdx.x;
    const int b   = bid >> 7;
    const int r   = bid & 127;
    const int h   = r >> 4;
    const int mb  = r & 15;
    const int n0  = mb * 16;

    const int tid  = threadIdx.x;
    const int w    = tid >> 5;
    const int l    = tid & 31;
    const int ln   = l & 15;
    const int half = l >> 4;
    const int c0   = w * 128;

    const float* arow = attn + ((size_t)((b * HH + h) * NN + n0) + ln) * TT;
    const u16*   xb   = xt + (size_t)b * DD * TT;

    f32x8 acc[8];
#pragma unroll
    for (int t = 0; t < 8; ++t) acc[t] = ZERO8;

    for (int ks = 0; ks < TT; ks += 32) {
        ABFrag a;
#pragma unroll
        for (int i = 0; i < 8; ++i) {
            const int k0 = ks + half * 16 + i * 2;
            const float2 f = *(const float2*)(arow + k0);
            a.u[i] = (unsigned)f2bf(f.x) | ((unsigned)f2bf(f.y) << 16);
        }
#pragma unroll
        for (int g = 0; g < 2; ++g) {
            ABFrag bf[4];
#pragma unroll
            for (int j = 0; j < 4; ++j) {
                const int n = c0 + (g * 4 + j) * 16 + ln;
                const u16* bp = xb + (size_t)n * TT + ks + half * 16;
#pragma unroll
                for (int i = 0; i < 8; ++i) bf[j].u[i] = *(const unsigned*)(bp + 2 * i);
            }
#pragma unroll
            for (int j = 0; j < 4; ++j)
                acc[g * 4 + j] = wmma_bf16(a, bf[j], acc[g * 4 + j]);
        }
    }

    // store: out layout (B, N, H, D) flattened
    float* ob = out + (size_t)b * (NN * HH * DD);
#pragma unroll
    for (int nt = 0; nt < 8; ++nt) {
        const int col = c0 + nt * 16 + ln;
#pragma unroll
        for (int r2 = 0; r2 < 8; ++r2)
            ob[(size_t)(n0 + half * 8 + r2) * (HH * DD) + h * DD + col] = acc[nt][r2];
    }
}

// ---------------------------------------------------------------------------
// Launch
// ---------------------------------------------------------------------------
extern "C" void kernel_launch(void* const* d_in, const int* in_sizes, int n_in,
                              void* d_out, int out_size, void* d_ws, size_t ws_size,
                              hipStream_t stream)
{
    (void)in_sizes; (void)n_in; (void)out_size; (void)ws_size;

    const float* q_in = (const float*)d_in[0];   // (B,N,D)
    const float* x_in = (const float*)d_in[1];   // (B,T,D)
    const float* gq   = (const float*)d_in[2];   // (D,)
    const float* gx   = (const float*)d_in[3];   // (D,)
    const float* wkv  = (const float*)d_in[4];   // (D, 2*DH)

    char* ws = (char*)d_ws;
    size_t off = 0;
    u16* xnb  = (u16*)(ws + off); off += (size_t)BB * TT * DD * 2;   // 134 MB
    u16* qnb  = (u16*)(ws + off); off += (size_t)BB * NN * DD * 2;   //   8 MB
    u16* wkt  = (u16*)(ws + off); off += (size_t)DH * DD * 2;        // 256 KB
    u16* kbuf = (u16*)(ws + off); off += (size_t)BB * TT * DH * 2;   //  16 MB
    u16* xt   = (u16*)(ws + off); off += (size_t)BB * DD * TT * 2;   // 134 MB

    float* out  = (float*)d_out;                             // (B, N*H*D)
    float* attn = out + (size_t)BB * NN * HH * DD;           // (B,H,N,T)

    ln_bf16_kernel<<<BB * TT, 256, 0, stream>>>(x_in, gx, xnb, 1.0f);
    ln_bf16_kernel<<<BB * NN, 256, 0, stream>>>(q_in, gq, qnb, SCALE);
    wkt_kernel<<<(DH * DD) / 256, 256, 0, stream>>>(wkv, wkt);
    xt_kernel<<<BB * (TT / 64) * (DD / 64), 256, 0, stream>>>(xnb, xt);
    kproj_kernel<<<BB * (TT / 128), 256, 0, stream>>>(xnb, wkt, kbuf);
    sim_softmax_kernel<<<BB * HH * (NN / 16), 1024, 0, stream>>>(qnb, kbuf, attn);
    out_gemm_kernel<<<BB * HH * (NN / 16), 256, 0, stream>>>(attn, xt, out);
}
